// ODEGPVAE_50543175139896
// MI455X (gfx1250) — matrix operand order repaired and checked by reading, not compiled, not run on hardware
//
#include <hip/hip_runtime.h>
#include <hip/hip_bf16.h>
#include <math.h>

typedef _Float16 h16;
typedef __attribute__((ext_vector_type(2)))  float v2f;
typedef __attribute__((ext_vector_type(8)))  float v8f;
typedef __attribute__((ext_vector_type(8)))  h16  v8h;
typedef __attribute__((ext_vector_type(16))) h16  v16h;

#define N_B   256
#define T_T   20
#define NF    16
#define Q     32
#define HDYN  256
#define HDIM  1024
#define DT    0.1f
#define LOG2PI 1.8378770664093453f
#define NTS   (N_B * T_T)          // 5120 decoded samples

// ---------------- WMMA wrappers (CDNA5 gfx1250) ----------------
__device__ __forceinline__ v8f wmma_f32_k4(v2f a, v2f b, v8f c) {
  // D = A(16x4 f32) * B(4x16 f32) + C   -> v_wmma_f32_16x16x4_f32
  return __builtin_amdgcn_wmma_f32_16x16x4_f32(false, a, false, b, (short)0, c, false, false);
}
__device__ __forceinline__ v8f wmma_f16_k32(v16h a, v16h b, v8f c) {
  // D = A(16x32 f16) * B(32x16 f16) + C -> v_wmma_f32_16x16x32_f16
  return __builtin_amdgcn_wmma_f32_16x16x32_f16(false, a, false, b, (short)0, c, false, false);
}
// Build a 16-wide f16 A fragment from two contiguous 8-element chunks.
__device__ __forceinline__ v16h pack16(v8h lo, v8h hi) {
  v16h a;
#pragma unroll
  for (int e = 0; e < 8; ++e) { a[e] = lo[e]; a[e + 8] = hi[e]; }
  return a;
}

// ---------------- small utility kernels ----------------
__global__ void init_scal(float* p) { if (threadIdx.x < 4) p[threadIdx.x] = 0.f; }

// f16 weight reorders: W2 -> W2t[n][k]; D1 -> [kykx][co][ci]; D2 -> [kykx][co][ci]
__global__ void cvt_f16(const float* __restrict__ W2, const float* __restrict__ D1,
                        const float* __restrict__ D2,
                        h16* __restrict__ W2t, h16* __restrict__ D1r, h16* __restrict__ D2r) {
  int gid = blockIdx.x * 256 + threadIdx.x;               // 49152 total
  if (gid < 8192) {
    int n = gid >> 8, k = gid & 255;
    W2t[gid] = (h16)W2[k * Q + n];
  } else if (gid < 8192 + 32768) {
    int t = gid - 8192;                 // [kk(16)][co(32)][ci(64)]
    int kk = t >> 11, co = (t >> 6) & 31, ci = t & 63;
    D1r[t] = (h16)D1[(co * 64 + ci) * 16 + kk];
  } else if (gid < 49152) {
    int t = gid - 40960;                // [kk(16)][co(16)][ci(32)]
    int kk = t >> 9, co = (t >> 5) & 15, ci = t & 31;
    D2r[t] = (h16)D2[(co * 32 + ci) * 16 + kk];
  }
}

// f32 weight transposes: W1t[n][k], FC heads -> [n][k], Wft[n][k], D3r[kk][ci]
__global__ void cvt_f32t(const float* __restrict__ W1,
                         const float* __restrict__ Wsm, const float* __restrict__ Wsl,
                         const float* __restrict__ Wvm, const float* __restrict__ Wvl,
                         const float* __restrict__ Wf,  const float* __restrict__ D3,
                         float* __restrict__ W1t, float* __restrict__ FCt,
                         float* __restrict__ Wft, float* __restrict__ D3r) {
  int gid = blockIdx.x * 256 + threadIdx.x;               // 180480 total
  if (gid < 16384) {                      // W1t[n(256)][k(64)]
    int n = gid >> 6, k = gid & 63;
    W1t[gid] = W1[k * HDYN + n];
  } else if (gid < 147456) {              // 4 x FC head transposes [n(32)][k(1024)]
    int t = gid - 16384;
    int fcid = t >> 15, u = t & 32767;
    int n = u >> 10, k = u & 1023;
    const float* src = (fcid == 0) ? Wsm : (fcid == 1) ? Wsl : (fcid == 2) ? Wvm : Wvl;
    FCt[(size_t)fcid * 32768 + u] = src[k * Q + n];
  } else if (gid < 180224) {              // Wft[n(1024)][k(32)]
    int u = gid - 147456;
    int n = u >> 5, k = u & 31;
    Wft[u] = Wf[k * HDIM + n];
  } else if (gid < 180480) {              // D3r[kk(16)][ci(16)]
    int u = gid - 180224;
    int kk = u >> 4, ci = u & 15;
    D3r[u] = D3[ci * 16 + kk];
  }
}

// ---------------- encoder conv stack (tiny; direct f32, LDS staged) ----------------
__global__ void enc_conv(const float* __restrict__ X, int cin,
                         const float* __restrict__ C1, const float* __restrict__ c1,
                         const float* __restrict__ C2, const float* __restrict__ c2,
                         const float* __restrict__ C3, const float* __restrict__ c3,
                         float* __restrict__ H) {
  __shared__ float h1[16 * 256];   // [16][16][16]
  __shared__ float h2[32 * 64];    // [32][8][8]
  __shared__ float h3[64 * 16];    // [64][4][4]
  const int n = blockIdx.x, tid = threadIdx.x;
  const float* x = X + (size_t)n * T_T * 1024;    // frames contiguous -> channels
  for (int o = tid; o < 16 * 256; o += blockDim.x) {
    int co = o >> 8, oy = (o >> 4) & 15, ox = o & 15;
    float acc = c1[co];
    for (int ci = 0; ci < cin; ++ci)
      for (int ky = 0; ky < 4; ++ky) {
        int iy = 2 * oy + ky - 1; if ((unsigned)iy >= 32u) continue;
        for (int kx = 0; kx < 4; ++kx) {
          int ix = 2 * ox + kx - 1; if ((unsigned)ix >= 32u) continue;
          acc += x[ci * 1024 + iy * 32 + ix] * C1[((co * cin + ci) * 4 + ky) * 4 + kx];
        }
      }
    h1[o] = fmaxf(acc, 0.f);
  }
  __syncthreads();
  for (int o = tid; o < 32 * 64; o += blockDim.x) {
    int co = o >> 6, oy = (o >> 3) & 7, ox = o & 7;
    float acc = c2[co];
    for (int ci = 0; ci < 16; ++ci)
      for (int ky = 0; ky < 4; ++ky) {
        int iy = 2 * oy + ky - 1; if ((unsigned)iy >= 16u) continue;
        for (int kx = 0; kx < 4; ++kx) {
          int ix = 2 * ox + kx - 1; if ((unsigned)ix >= 16u) continue;
          acc += h1[ci * 256 + iy * 16 + ix] * C2[((co * 16 + ci) * 4 + ky) * 4 + kx];
        }
      }
    h2[o] = fmaxf(acc, 0.f);
  }
  __syncthreads();
  for (int o = tid; o < 64 * 16; o += blockDim.x) {
    int co = o >> 4, oy = (o >> 2) & 3, ox = o & 3;
    float acc = c3[co];
    for (int ci = 0; ci < 32; ++ci)
      for (int ky = 0; ky < 4; ++ky) {
        int iy = 2 * oy + ky - 1; if ((unsigned)iy >= 8u) continue;
        for (int kx = 0; kx < 4; ++kx) {
          int ix = 2 * ox + kx - 1; if ((unsigned)ix >= 8u) continue;
          acc += h2[ci * 64 + iy * 8 + ix] * C3[((co * 32 + ci) * 4 + ky) * 4 + kx];
        }
      }
    h3[o] = fmaxf(acc, 0.f);
  }
  __syncthreads();
  for (int o = tid; o < HDIM; o += blockDim.x) H[(size_t)n * HDIM + o] = h3[o];
}

// ---------------- encoder FC heads: [256,1024]x[1024,32] via f32 WMMA ----------------
__global__ void enc_fc(const float* __restrict__ Hs, const float* __restrict__ Hv,
                       const float* __restrict__ FCt,
                       const float* __restrict__ bsm, const float* __restrict__ bsl,
                       const float* __restrict__ bvm, const float* __restrict__ bvl,
                       float* __restrict__ MUs, float* __restrict__ LVs,
                       float* __restrict__ MUv, float* __restrict__ LVv) {
  int b = blockIdx.x;                      // 128 = 4 fc * 16 mtiles * 2 ntiles
  int fc = b & 3, mt = (b >> 2) & 15, nt = (b >> 6) & 1;
  const float* H = (fc < 2) ? Hs : Hv;
  const float* Wt = FCt + (size_t)fc * 32768;     // [n(32)][k(1024)]
  const float* bias; float* dst;
  if (fc == 0)      { bias = bsm; dst = MUs; }
  else if (fc == 1) { bias = bsl; dst = LVs; }
  else if (fc == 2) { bias = bvm; dst = MUv; }
  else              { bias = bvl; dst = LVv; }
  int lane = threadIdx.x & 31, r = lane & 15, g = lane >> 4;
  __builtin_prefetch(Wt, 0, 0);
  v8f acc = {};
  for (int kb = 0; kb < HDIM; kb += 4) {
    v2f a  = *(const v2f*)&H[(size_t)(mt * 16 + r) * HDIM + kb + 2 * g];
    v2f bb = *(const v2f*)&Wt[(size_t)(nt * 16 + r) * HDIM + kb + 2 * g];
    acc = wmma_f32_k4(a, bb, acc);
  }
#pragma unroll
  for (int v = 0; v < 8; ++v) {
    int m = v + 8 * g;
    dst[(mt * 16 + m) * Q + nt * 16 + r] = acc[v] + bias[nt * 16 + r];
  }
}

// ---------------- sampler: z0, qz0_m/logv, logp0 ----------------
__global__ void sampler(const float* __restrict__ MUs, const float* __restrict__ LVs,
                        const float* __restrict__ MUv, const float* __restrict__ LVv,
                        const float* __restrict__ eps_s, const float* __restrict__ eps_v,
                        float* __restrict__ qm, float* __restrict__ qlv,
                        float* __restrict__ zt, float* __restrict__ logp) {
  __shared__ float red[64];
  int n = blockIdx.x, j = threadIdx.x;     // 64 threads
  float mu, lv, eps;
  if (j < Q) { mu = MUv[n * Q + j]; lv = LVv[n * Q + j]; eps = eps_v[n * Q + j]; }
  else { int j2 = j - Q; mu = MUs[n * Q + j2]; lv = LVs[n * Q + j2]; eps = eps_s[n * Q + j2]; }
  float val = mu + eps * __expf(lv);
  qm[n * 64 + j]  = mu;
  qlv[n * 64 + j] = lv;
  zt[((size_t)n * T_T) * 64 + j] = val;
  red[j] = eps * eps;
  __syncthreads();
  for (int s = 32; s > 0; s >>= 1) { if (j < s) red[j] += red[j + s]; __syncthreads(); }
  if (j == 0) logp[n * T_T] = -0.5f * red[0] - (float)Q * LOG2PI;
}

// ---------------- RK4 ODE: per-block independent 16-row slice, all-LDS ----------------
__global__ void ode_rk4(const float* __restrict__ W1, const float* __restrict__ W1t,
                        const float* __restrict__ b1,
                        const float* __restrict__ W2f, const h16* __restrict__ W2t,
                        const float* __restrict__ b2,
                        float* __restrict__ zt, float* __restrict__ logp) {
  __shared__ float z[16][64], vs[16][64], kbuf[16][64], accb[16][64];
  __shared__ h16 hbuf[16][HDYN];
  __shared__ h16 sW2[Q * HDYN];            // W2t staged: [n(32)][k(256)], 16 KB
  __shared__ float cvec[HDYN];
  __shared__ float tr[16], lp[16], lpacc[16];
  const int tid = threadIdx.x, lane = tid & 31, wave = tid >> 5;   // 128 thr, 4 waves
  const int r = lane & 15, g = lane >> 4;
  const int r0 = blockIdx.x * 16;
  __builtin_prefetch(W1t, 0, 0);
  // stage W2t into LDS (b128 chunks)
  for (int o = tid; o < (Q * HDYN) / 8; o += blockDim.x)
    ((v8h*)sW2)[o] = ((const v8h*)W2t)[o];
  // c_k = sum_{i<Q} W1[i,k] * W2[k,i]
  for (int k = tid; k < HDYN; k += blockDim.x) {
    float s = 0.f;
    for (int i = 0; i < Q; ++i) s += W1[i * HDYN + k] * W2f[k * Q + i];
    cvec[k] = s;
  }
  for (int o = tid; o < 16 * 64; o += blockDim.x)
    z[o >> 6][o & 63] = zt[((size_t)(r0 + (o >> 6)) * T_T) * 64 + (o & 63)];
  if (tid < 16) lp[tid] = logp[(r0 + tid) * T_T];
  __syncthreads();

  for (int step = 0; step < T_T - 1; ++step) {
    if (tid < 16) lpacc[tid] = 0.f;
    for (int ev = 0; ev < 4; ++ev) {
      const float a = (ev == 0) ? 0.f : ((ev == 3) ? DT : 0.5f * DT);
      const float w = (ev == 0 || ev == 3) ? 1.f : 2.f;
      for (int o = tid; o < 16 * 64; o += blockDim.x) {
        int rr = o >> 6, cc = o & 63;
        vs[rr][cc] = (ev == 0) ? z[rr][cc] : z[rr][cc] + a * kbuf[rr][cc];
      }
      if (tid < 16) tr[tid] = 0.f;
      __syncthreads();
      // GEMM1: [16,64] x [64,256] (f32 WMMA), tanh -> hbuf(f16), trace partials
      float trp[8];
#pragma unroll
      for (int v = 0; v < 8; ++v) trp[v] = 0.f;
      for (int i = 0; i < 4; ++i) {
        int nt = wave * 4 + i;
        v8f acc = {};
#pragma unroll
        for (int kc = 0; kc < 16; ++kc) {
          v2f av = *(const v2f*)&vs[r][kc * 4 + 2 * g];
          v2f bv = *(const v2f*)&W1t[(size_t)(nt * 16 + r) * 64 + kc * 4 + 2 * g];
          acc = wmma_f32_k4(av, bv, acc);
        }
#pragma unroll
        for (int v = 0; v < 8; ++v) {
          int m = v + 8 * g;
          float hv = tanhf(acc[v] + b1[nt * 16 + r]);
          hbuf[m][nt * 16 + r] = (h16)hv;
          trp[v] += (1.f - hv * hv) * cvec[nt * 16 + r];
        }
      }
#pragma unroll
      for (int v = 0; v < 8; ++v) atomicAdd(&tr[v + 8 * g], trp[v]);
      __syncthreads();
      // GEMM2: [16,256](f16) x [256,32](f16) -> dv ; plus dvs tail copy
      if (wave < 2) {
        int nt = wave;
        v8f acc = {};
        for (int kc = 0; kc < 8; ++kc) {
          v8h alo = *(const v8h*)&hbuf[r][kc * 32 + 16 * g];
          v8h ahi = *(const v8h*)&hbuf[r][kc * 32 + 16 * g + 8];
          v16h bh = *(const v16h*)&sW2[(nt * 16 + r) * HDYN + kc * 32 + 16 * g];
          acc = wmma_f16_k32(pack16(alo, ahi), bh, acc);
        }
#pragma unroll
        for (int v = 0; v < 8; ++v) {
          int m = v + 8 * g;
          kbuf[m][nt * 16 + r] = acc[v] + b2[nt * 16 + r];
        }
      } else {
        int t2 = tid - 64;
        for (int o = t2; o < 16 * Q; o += 64) {
          int rr = o >> 5, cc = o & 31;
          kbuf[rr][Q + cc] = vs[rr][cc];   // dvs = [dv, vs[:, :Q]]
        }
      }
      __syncthreads();
      for (int o = tid; o < 16 * 64; o += blockDim.x) {
        int rr = o >> 6, cc = o & 63;
        accb[rr][cc] = (ev == 0) ? kbuf[rr][cc] : accb[rr][cc] + w * kbuf[rr][cc];
      }
      if (tid < 16) lpacc[tid] += w * (-tr[tid]);
      __syncthreads();
    }
    for (int o = tid; o < 16 * 64; o += blockDim.x) {
      int rr = o >> 6, cc = o & 63;
      float zn = z[rr][cc] + (DT / 6.f) * accb[rr][cc];
      z[rr][cc] = zn;
      zt[((size_t)(r0 + rr) * T_T + step + 1) * 64 + cc] = zn;
    }
    if (tid < 16) {
      lp[tid] += (DT / 6.f) * lpacc[tid];
      logp[(r0 + tid) * T_T + step + 1] = lp[tid];
    }
    __syncthreads();
  }
}

// ------- fc3: s[5120,32] x Wf[32,1024] (f32 WMMA) -> f16, layout [s][pix16][ch64] -------
__global__ void fc3_gemm(const float* __restrict__ zt, const float* __restrict__ Wft,
                         const float* __restrict__ bf, h16* __restrict__ out) {
  int wave = threadIdx.x >> 5, lane = threadIdx.x & 31;
  int tile = blockIdx.x * 8 + wave;            // 20480 tiles (320 M x 64 N)
  int mt = tile >> 6, nt = tile & 63;          // col = nt*16+r -> ch = nt, pix = r
  int r = lane & 15, g = lane >> 4;
  __builtin_prefetch(Wft, 0, 0);
  v8f acc = {};
#pragma unroll
  for (int kb = 0; kb < 8; ++kb) {
    int k0 = kb * 4 + 2 * g;
    v2f a = *(const v2f*)&zt[(size_t)(mt * 16 + r) * 64 + Q + k0];
    v2f b = *(const v2f*)&Wft[(size_t)(nt * 16 + r) * Q + k0];
    acc = wmma_f32_k4(a, b, acc);
  }
#pragma unroll
  for (int v = 0; v < 8; ++v) {
    int m = v + 8 * g;
    out[(size_t)(mt * 16 + m) * HDIM + r * 64 + nt] = (h16)(acc[v] + bf[nt * 16 + r]);
  }
}

// ---- deconv1: 64ch 4x4 -> 32ch 8x8, implicit GEMM f16 WMMA, out [s][pix64][ch32] ----
__global__ void deconv1(const h16* __restrict__ xin, const h16* __restrict__ Wr,
                        const float* __restrict__ bias, h16* __restrict__ out) {
  int mg = blockIdx.x >> 6, pix = blockIdx.x & 63;
  int oy = pix >> 3, ox = pix & 7;
  int nt = threadIdx.x >> 5, lane = threadIdx.x & 31;   // 64 thr, 2 N-tiles
  int r = lane & 15, g = lane >> 4;
  __builtin_prefetch(Wr, 0, 0);
  v8f acc = {};
  int k0y = (oy + 1) & 1, k0x = (ox + 1) & 1;
  for (int ky = k0y; ky < 4; ky += 2) {
    int iy = (oy + 1 - ky) >> 1; if ((unsigned)iy >= 4u) continue;
    for (int kx = k0x; kx < 4; kx += 2) {
      int ix = (ox + 1 - kx) >> 1; if ((unsigned)ix >= 4u) continue;
      int pixIn = iy * 4 + ix, kk = ky * 4 + kx;
      const h16* ab = xin + (size_t)(mg * 16 + r) * 1024 + pixIn * 64;
#pragma unroll
      for (int kc = 0; kc < 2; ++kc) {
        v8h alo = *(const v8h*)(ab + kc * 32 + 16 * g);
        v8h ahi = *(const v8h*)(ab + kc * 32 + 16 * g + 8);
        v16h bh = *(const v16h*)&Wr[((size_t)kk * 32 + nt * 16 + r) * 64 + kc * 32 + 16 * g];
        acc = wmma_f16_k32(pack16(alo, ahi), bh, acc);
      }
    }
  }
#pragma unroll
  for (int v = 0; v < 8; ++v) {
    int m = v + 8 * g;
    float val = fmaxf(acc[v] + bias[nt * 16 + r], 0.f);
    out[(size_t)(mg * 16 + m) * 2048 + (oy * 8 + ox) * 32 + nt * 16 + r] = (h16)val;
  }
}

// ---- deconv2: 32ch 8x8 -> 16ch 16x16, implicit GEMM f16 WMMA, out [s][pix256][ch16] ----
__global__ void deconv2k(const h16* __restrict__ xin, const h16* __restrict__ Wr,
                         const float* __restrict__ bias, h16* __restrict__ out) {
  int mg = blockIdx.x >> 8, pix = blockIdx.x & 255;
  int oy = pix >> 4, ox = pix & 15;
  int lane = threadIdx.x & 31, r = lane & 15, g = lane >> 4;
  __builtin_prefetch(Wr, 0, 0);
  v8f acc = {};
  int k0y = (oy + 1) & 1, k0x = (ox + 1) & 1;
  for (int ky = k0y; ky < 4; ky += 2) {
    int iy = (oy + 1 - ky) >> 1; if ((unsigned)iy >= 8u) continue;
    for (int kx = k0x; kx < 4; kx += 2) {
      int ix = (ox + 1 - kx) >> 1; if ((unsigned)ix >= 8u) continue;
      int pixIn = iy * 8 + ix, kk = ky * 4 + kx;
      const h16* ab = xin + (size_t)(mg * 16 + r) * 2048 + pixIn * 32;
      v8h alo = *(const v8h*)(ab + 16 * g);
      v8h ahi = *(const v8h*)(ab + 16 * g + 8);
      v16h bh = *(const v16h*)&Wr[((size_t)kk * 16 + r) * 32 + 16 * g];
      acc = wmma_f16_k32(pack16(alo, ahi), bh, acc);
    }
  }
#pragma unroll
  for (int v = 0; v < 8; ++v) {
    int m = v + 8 * g;
    float val = fmaxf(acc[v] + bias[r], 0.f);
    out[(size_t)(mg * 16 + m) * 4096 + (oy * 16 + ox) * 16 + r] = (h16)val;
  }
}

// ---------------- deconv3 (16ch->1ch) + sigmoid + Bernoulli log-likelihood ----------------
__global__ void deconv3_lh(const h16* __restrict__ xin, const float* __restrict__ Wr,
                           const float* __restrict__ bias, const float* __restrict__ X,
                           float* __restrict__ Xrec, float* __restrict__ lh_sum) {
  __shared__ float red[256];
  size_t gid = (size_t)blockIdx.x * blockDim.x + threadIdx.x;   // exactly 5,242,880
  int s = (int)(gid >> 10);
  int p = (int)(gid & 1023);
  int oy = p >> 5, ox = p & 31;
  float acc = bias[0];
  int k0y = (oy + 1) & 1, k0x = (ox + 1) & 1;
  for (int ky = k0y; ky < 4; ky += 2) {
    int iy = (oy + 1 - ky) >> 1; if ((unsigned)iy >= 16u) continue;
    for (int kx = k0x; kx < 4; kx += 2) {
      int ix = (ox + 1 - kx) >> 1; if ((unsigned)ix >= 16u) continue;
      int pixIn = iy * 16 + ix, kk = ky * 4 + kx;
      v8h lo = *(const v8h*)(xin + (size_t)s * 4096 + pixIn * 16);
      v8h hi = *(const v8h*)(xin + (size_t)s * 4096 + pixIn * 16 + 8);
#pragma unroll
      for (int ci = 0; ci < 8; ++ci) {
        acc += (float)lo[ci] * Wr[kk * 16 + ci];
        acc += (float)hi[ci] * Wr[kk * 16 + 8 + ci];
      }
    }
  }
  float xr = 1.f / (1.f + __expf(-acc));
  Xrec[gid] = xr;
  float x = X[gid];
  float c = __logf(0.001f + xr) * x + __logf(1.001f - xr) * (1.f - x);
  red[threadIdx.x] = c;
  __syncthreads();
  for (int st = 128; st > 0; st >>= 1) {
    if (threadIdx.x < st) red[threadIdx.x] += red[threadIdx.x + st];
    __syncthreads();
  }
  if (threadIdx.x == 0) atomicAdd(lh_sum, red[0]);
}

// ---------------- KL(z) reduction ----------------
__global__ void klz_red(const float* __restrict__ zt, const float* __restrict__ logp,
                        float* __restrict__ klz_sum) {
  __shared__ float red[32];
  int n = blockIdx.x, t = threadIdx.x;  // 32 threads
  float term = 0.f;
  if (t < T_T) {
    const float* zp = zt + ((size_t)n * T_T + t) * 64;
    float sq = 0.f;
    for (int d = 0; d < 64; ++d) sq += zp[d] * zp[d];
    float log_pzt = -0.5f * sq - (float)Q * LOG2PI;
    term = logp[n * T_T + t] - log_pzt;
  }
  red[t] = term;
  __syncthreads();
  for (int s = 16; s > 0; s >>= 1) { if (t < s) red[t] += red[t + s]; __syncthreads(); }
  if (t == 0) atomicAdd(klz_sum, red[0]);
}

// ---------------- KL(w) = 0.5(|W1|^2 + |W2|^2) ----------------
__global__ void klw_red(const float* __restrict__ W1, const float* __restrict__ W2,
                        float* __restrict__ klw_sum) {
  __shared__ float red[256];
  int gid = blockIdx.x * 256 + threadIdx.x;   // 24576 = 16384 + 8192
  float v = (gid < 2 * Q * HDYN) ? W1[gid] : W2[gid - 2 * Q * HDYN];
  red[threadIdx.x] = v * v;
  __syncthreads();
  for (int st = 128; st > 0; st >>= 1) {
    if (threadIdx.x < st) red[threadIdx.x] += red[threadIdx.x + st];
    __syncthreads();
  }
  if (threadIdx.x == 0) atomicAdd(klw_sum, red[0]);
}

// ---------------- finalize scalars ----------------
__global__ void finalize(const float* __restrict__ scal, const int* __restrict__ Ndata,
                         float* __restrict__ out_scalars) {
  float Nd  = (float)Ndata[0];
  float lh  = Nd * scal[0] / (float)N_B;
  float klz = Nd * scal[1] / (float)N_B;
  float klw = 0.5f * scal[2];
  out_scalars[0] = lh - klz - 1.0f * klw;   // elbo (BETA = 1)
  out_scalars[1] = lh;
  out_scalars[2] = klz;
  out_scalars[3] = klw;                     // BETA * kl_w
}

// ======================= launch =======================
extern "C" void kernel_launch(void* const* d_in, const int* in_sizes, int n_in,
                              void* d_out, int out_size, void* d_ws, size_t ws_size,
                              hipStream_t stream) {
  const float* X     = (const float*)d_in[0];
  const float* eps_s = (const float*)d_in[1];
  const float* eps_v = (const float*)d_in[2];
  const float* Cs1 = (const float*)d_in[3];  const float* cs1 = (const float*)d_in[4];
  const float* Cs2 = (const float*)d_in[5];  const float* cs2 = (const float*)d_in[6];
  const float* Cs3 = (const float*)d_in[7];  const float* cs3 = (const float*)d_in[8];
  const float* Wsm = (const float*)d_in[9];  const float* bsm = (const float*)d_in[10];
  const float* Wsl = (const float*)d_in[11]; const float* bsl = (const float*)d_in[12];
  const float* Cv1 = (const float*)d_in[13]; const float* cv1 = (const float*)d_in[14];
  const float* Cv2 = (const float*)d_in[15]; const float* cv2 = (const float*)d_in[16];
  const float* Cv3 = (const float*)d_in[17]; const float* cv3 = (const float*)d_in[18];
  const float* Wvm = (const float*)d_in[19]; const float* bvm = (const float*)d_in[20];
  const float* Wvl = (const float*)d_in[21]; const float* bvl = (const float*)d_in[22];
  const float* W1  = (const float*)d_in[23]; const float* b1  = (const float*)d_in[24];
  const float* W2  = (const float*)d_in[25]; const float* b2  = (const float*)d_in[26];
  const float* Wf  = (const float*)d_in[27]; const float* bf  = (const float*)d_in[28];
  const float* D1  = (const float*)d_in[29]; const float* d1  = (const float*)d_in[30];
  const float* D2  = (const float*)d_in[31]; const float* d2  = (const float*)d_in[32];
  const float* D3  = (const float*)d_in[33]; const float* d3  = (const float*)d_in[34];
  const int*   Ndp = (const int*)d_in[35];

  // workspace arena
  char* ws = (char*)d_ws;
  size_t off = 0;
  auto arena = [&](size_t bytes) -> void* {
    void* p = ws + off;
    off += (bytes + 255) & ~(size_t)255;
    return p;
  };
  float* Hs   = (float*)arena((size_t)N_B * HDIM * 4);
  float* Hv   = (float*)arena((size_t)N_B * HDIM * 4);
  float* MUs  = (float*)arena((size_t)N_B * Q * 4);
  float* LVs  = (float*)arena((size_t)N_B * Q * 4);
  float* MUv  = (float*)arena((size_t)N_B * Q * 4);
  float* LVv  = (float*)arena((size_t)N_B * Q * 4);
  float* logp = (float*)arena((size_t)N_B * T_T * 4);
  float* scal = (float*)arena(256);                 // [lh_sum, klz_sum, klw_sum]
  float* W1t  = (float*)arena((size_t)HDYN * 64 * 4);       // [n(256)][k(64)]
  float* FCt  = (float*)arena((size_t)4 * Q * HDIM * 4);    // 4 x [n(32)][k(1024)]
  float* Wft  = (float*)arena((size_t)HDIM * Q * 4);        // [n(1024)][k(32)]
  float* D3r  = (float*)arena((size_t)16 * 16 * 4);         // [kk(16)][ci(16)]
  h16* W2t  = (h16*)arena((size_t)Q * HDYN * 2);            // [n(32)][k(256)]
  h16* D1r  = (h16*)arena((size_t)16 * 32 * 64 * 2);        // [kk][co][ci]
  h16* D2r  = (h16*)arena((size_t)16 * 16 * 32 * 2);        // [kk][co][ci]
  h16* fc3h = (h16*)arena((size_t)NTS * HDIM * 2);          // [s][pix16][ch64]
  h16* dc1h = (h16*)arena((size_t)NTS * 2048 * 2);          // [s][pix64][ch32]
  h16* dc2h = (h16*)arena((size_t)NTS * 4096 * 2);          // [s][pix256][ch16]
  (void)ws_size; (void)n_in; (void)in_sizes; (void)out_size;

  // output layout (concatenated, reference return order)
  float* outf  = (float*)d_out;
  float* Xrec  = outf;                              // 1*256*20*1*32*32
  float* qm    = outf + (size_t)5242880;            // 256*64
  float* qlv   = qm + 16384;                        // 256*64
  float* zt    = qlv + 16384;                       // 1*256*20*64
  float* oscal = zt + 327680;                       // elbo, lh, klz, beta*klw

  hipLaunchKernelGGL(init_scal, dim3(1), dim3(32), 0, stream, scal);
  hipLaunchKernelGGL(cvt_f16, dim3(192), dim3(256), 0, stream, W2, D1, D2, W2t, D1r, D2r);
  hipLaunchKernelGGL(cvt_f32t, dim3(705), dim3(256), 0, stream,
                     W1, Wsm, Wsl, Wvm, Wvl, Wf, D3, W1t, FCt, Wft, D3r);
  hipLaunchKernelGGL(enc_conv, dim3(N_B), dim3(256), 0, stream,
                     X, 1, Cs1, cs1, Cs2, cs2, Cs3, cs3, Hs);
  hipLaunchKernelGGL(enc_conv, dim3(N_B), dim3(256), 0, stream,
                     X, 10, Cv1, cv1, Cv2, cv2, Cv3, cv3, Hv);
  hipLaunchKernelGGL(enc_fc, dim3(128), dim3(32), 0, stream,
                     Hs, Hv, FCt, bsm, bsl, bvm, bvl, MUs, LVs, MUv, LVv);
  hipLaunchKernelGGL(sampler, dim3(N_B), dim3(64), 0, stream,
                     MUs, LVs, MUv, LVv, eps_s, eps_v, qm, qlv, zt, logp);
  hipLaunchKernelGGL(ode_rk4, dim3(16), dim3(128), 0, stream,
                     W1, W1t, b1, W2, W2t, b2, zt, logp);
  hipLaunchKernelGGL(fc3_gemm, dim3(2560), dim3(256), 0, stream, zt, Wft, bf, fc3h);
  hipLaunchKernelGGL(deconv1, dim3(320 * 64), dim3(64), 0, stream, fc3h, D1r, d1, dc1h);
  hipLaunchKernelGGL(deconv2k, dim3(320 * 256), dim3(32), 0, stream, dc1h, D2r, d2, dc2h);
  hipLaunchKernelGGL(deconv3_lh, dim3(20480), dim3(256), 0, stream,
                     dc2h, D3r, d3, X, Xrec, &scal[0]);
  hipLaunchKernelGGL(klz_red, dim3(N_B), dim3(32), 0, stream, zt, logp, &scal[1]);
  hipLaunchKernelGGL(klw_red, dim3(96), dim3(256), 0, stream, W1, W2, &scal[2]);
  hipLaunchKernelGGL(finalize, dim3(1), dim3(1), 0, stream, scal, Ndp, oscal);
}
